// QuadraticHeterogenousCrossAttention_41764261986703
// MI455X (gfx1250) — compile-verified, matrix operand order checked by reading, not run on hardware
//
#include <hip/hip_runtime.h>
#include <hip/hip_bf16.h>

// Problem constants (match reference)
#define B_   2
#define S_   512
#define D_   512
#define H_   8
#define DH_  64
#define BS_  (B_ * S_)          // 1024

typedef __attribute__((ext_vector_type(16))) _Float16 v16h;
typedef __attribute__((ext_vector_type(8)))  _Float16 v8h;   // 16-byte unit
typedef __attribute__((ext_vector_type(8)))  float    v8f;

__device__ __forceinline__ v8f wmma_f32_f16(v16h a, v16h b, v8f c) {
  // D = A(16x32 f16) * B(32x16 f16) + C(16x16 f32)
  return __builtin_amdgcn_wmma_f32_16x16x32_f16(false, a, false, b, (short)0, c,
                                                false, false);
}

// A-layout loader (16x32 f16 per ISA 7.12.2). Also loads B-tiles when the
// source matrix is stored transposed (B-layout striping == A-layout on M^T).
//   lane<16 : row = lane,     K slots {k0+0..7, k0+16..23}
//   lane>=16: row = lane-16,  K slots {k0+8..15, k0+24..31}
// Per lane: two contiguous 16B chunks -> 2x b128.
__device__ __forceinline__ v16h load_a_f16(const _Float16* A, int lda,
                                           int row0, int k0, int lane) {
  const int half = lane >> 4;
  const int r = lane & 15;
  const _Float16* p = A + (size_t)(row0 + r) * lda + k0 + half * 8;
  v16h o;
#pragma unroll
  for (int i = 0; i < 8; ++i) o[i] = p[i];
#pragma unroll
  for (int i = 0; i < 8; ++i) o[8 + i] = p[16 + i];
  return o;
}

// ---- 0a) one-time f32 -> f16 conversion ------------------------------------
__global__ void f32_to_f16_kernel(const float* __restrict__ src,
                                  _Float16* __restrict__ dst, int n) {
  int idx = blockIdx.x * blockDim.x + threadIdx.x;
  if (idx < n) dst[idx] = (_Float16)src[idx];
}

// ---- 0b) one-time f32 -> f16 transpose (512x512): dst[n][k] = src[k][n] ----
__global__ void transpose_conv_kernel(const float* __restrict__ src,
                                      _Float16* __restrict__ dst) {
  int idx = blockIdx.x * blockDim.x + threadIdx.x;
  int k = idx >> 9, n = idx & (D_ - 1);
  dst[(size_t)n * D_ + k] = (_Float16)src[idx];
}

// ---- 1) kv = softmax(feature_weights) . kv_feats  (emit f16) ---------------
__global__ void kv_combine_kernel(const float* __restrict__ kvf,
                                  const float* __restrict__ fw,
                                  _Float16* __restrict__ kv) {
  const int idx = blockIdx.x * blockDim.x + threadIdx.x;   // exact coverage
  float w0 = fw[0], w1 = fw[1], w2 = fw[2], w3 = fw[3];
  float mx = fmaxf(fmaxf(w0, w1), fmaxf(w2, w3));
  float e0 = expf(w0 - mx), e1 = expf(w1 - mx);
  float e2 = expf(w2 - mx), e3 = expf(w3 - mx);
  float inv = 1.0f / (e0 + e1 + e2 + e3);
  const int n = BS_ * D_;
  kv[idx] = (_Float16)((e0 * kvf[idx] + e1 * kvf[n + idx] +
                        e2 * kvf[2 * n + idx] + e3 * kvf[3 * n + idx]) * inv);
}

// ---- 2) Q/K/V projections (WMMA, transposed weights) + RoPE ----------------
// One wave = one 16x16 output tile. No LDS, no barriers.
__global__ void proj_qkv_kernel(const _Float16* __restrict__ q16,
                                const _Float16* __restrict__ kv16,
                                const _Float16* __restrict__ WqT, const float* __restrict__ bq,
                                const _Float16* __restrict__ WkT, const float* __restrict__ bk,
                                const _Float16* __restrict__ WvT, const float* __restrict__ bv,
                                const float* __restrict__ offset,
                                _Float16* __restrict__ qf,
                                _Float16* __restrict__ kf,
                                _Float16* __restrict__ vT) {
  const int tid  = threadIdx.x;
  const int lane = tid & 31;
  const int w    = tid >> 5;
  const int half = lane >> 4;
  const int c16  = lane & 15;
  const int sel  = blockIdx.z;            // 0=q, 1=k, 2=v
  const int col0 = (blockIdx.x * 4 + w) * 16;
  const int row0 = blockIdx.y * 16;

  const _Float16* A  = (sel == 0) ? q16 : kv16;
  const _Float16* WT = (sel == 0) ? WqT : (sel == 1) ? WkT : WvT;
  const float* bvec  = (sel == 0) ? bq : (sel == 1) ? bk : bv;
  const int col = col0 + c16;

  v8f acc;
  {
    float bb = bvec[col];
#pragma unroll
    for (int r = 0; r < 8; ++r) acc[r] = bb;
  }
#pragma unroll 4
  for (int k0 = 0; k0 < D_; k0 += 32) {
    v16h a = load_a_f16(A,  D_, row0, k0, lane);
    v16h b = load_a_f16(WT, D_, col0, k0, lane);   // WT[n][k]: B-tile via A-load
    acc = wmma_f32_f16(a, b, acc);
    if (k0 + 32 < D_)
      __builtin_prefetch(&WT[(size_t)col0 * D_ + k0 + 32], 0, 1);
  }

  const int h = col >> 6;       // head
  const int c = col & 63;       // index within head
  const int mbase = row0 + 8 * half;
  const int bidx  = mbase >> 9;           // batch (constant over tile rows)
  const int sbase = mbase & (S_ - 1);

  if (sel == 2) {
    // V stored transposed: vT[b][h][dh][s]; 8 consecutive s -> one 16B store
    v8h pv;
#pragma unroll
    for (int r = 0; r < 8; ++r) pv[r] = (_Float16)acc[r];
    *(v8h*)&vT[((size_t)(bidx * H_ + h) * DH_ + c) * S_ + sbase] = pv;
    return;
  }

  float offh = offset[h];
  int pairi = c >> 1;
  float freq = powf(1000.0f, -2.0f * (float)pairi / (float)DH_);
  const int odd = c & 1;
  _Float16* out = (sel == 0) ? qf : kf;
#pragma unroll
  for (int r = 0; r < 8; ++r) {
    float val = acc[r];
    float par = __shfl_xor(val, 1);       // RoPE partner column (col ^ 1)
    int s = sbase + r;
    int pos = s >> 6;                     // s // N
    float ang = ((float)pos + offh) * freq;
    float cv = cosf(ang), sv = sinf(ang);
    val = odd ? (par * sv + val * cv) : (val * cv - par * sv);
    if (sel == 0) val *= 0.125f;          // fold 1/sqrt(DH) into q
    out[((size_t)(bidx * H_ + h) * S_ + s) * DH_ + c] = (_Float16)val;
  }
}

// ---- 3) SH bias MLP: 9 -> 64 -> 64 -> H, all WMMA --------------------------
// Weights held transposed in LDS so every B-tile is 2x contiguous ds_load_b128.
__global__ void sh_bias_kernel(const float* __restrict__ coords,
                               const float* __restrict__ W1, const float* __restrict__ b1,
                               const float* __restrict__ W2, const float* __restrict__ b2,
                               const float* __restrict__ W3, const float* __restrict__ b3,
                               float* __restrict__ bias) {
  __shared__ _Float16 sW1t[64 * 32];     // [n][k], k padded 9->32 (zeros)
  __shared__ _Float16 sW2t[64 * 64];     // [n][k]
  __shared__ _Float16 sW3t[16 * 64];     // [n][k], n padded 8->16 (zeros)
  __shared__ _Float16 sT[4][16 * 64];    // per-wave D-layout -> A-layout bounce

  const int tid = threadIdx.x;
  for (int e = tid; e < 64 * 32; e += 128) {
    int n = e >> 5, k = e & 31;
    sW1t[e] = (_Float16)((k < 9) ? W1[k * 64 + n] : 0.0f);
  }
  for (int e = tid; e < 64 * 64; e += 128) {
    int n = e >> 6, k = e & 63;
    sW2t[e] = (_Float16)W2[k * 64 + n];
  }
  for (int e = tid; e < 16 * 64; e += 128) {
    int n = e >> 6, k = e & 63;
    sW3t[e] = (_Float16)((n < 8) ? W3[k * 8 + n] : 0.0f);
  }
  __syncthreads();

  const int lane = tid & 31;
  const int w    = tid >> 5;
  const int half = lane >> 4;
  const int c16  = lane & 15;

  const int b = blockIdx.x / S_;
  const int i = blockIdx.x - b * S_;

  const float cix = coords[(size_t)(b * S_ + i) * 3 + 0];
  const float ciy = coords[(size_t)(b * S_ + i) * 3 + 1];
  const float ciz = coords[(size_t)(b * S_ + i) * 3 + 2];

  _Float16* tp = &sT[w][0];

  for (int jt = w; jt < S_ / 16; jt += 4) {
    const int j0 = jt * 16;
    const int j  = j0 + c16;               // pair row (both lane halves)
    float rx = cix - coords[(size_t)(b * S_ + j) * 3 + 0];
    float ry = ciy - coords[(size_t)(b * S_ + j) * 3 + 1];
    float rz = ciz - coords[(size_t)(b * S_ + j) * 3 + 2];
    float nrm = sqrtf(rx * rx + ry * ry + rz * rz) + 1e-6f;
    float x = rx / nrm, y = ry / nrm, z = rz / nrm;

    const float c0  = 0.28209479177387814f;
    const float c1  = 0.4886025119029199f;
    const float c2a = 1.0925484305920792f;
    const float c2b = 0.31539156525252005f;
    const float c2c = 0.5462742152960396f;

    v16h aSh;
#pragma unroll
    for (int t = 0; t < 16; ++t) aSh[t] = (_Float16)0.0f;
    if (half == 0) {                       // K slots 0..7 = feats 0..7
      aSh[0] = (_Float16)c0;
      aSh[1] = (_Float16)(c1 * x);
      aSh[2] = (_Float16)(c1 * y);
      aSh[3] = (_Float16)(c1 * z);
      aSh[4] = (_Float16)(c2a * x * y);
      aSh[5] = (_Float16)(c2a * y * z);
      aSh[6] = (_Float16)(c2b * (3.0f * z * z - 1.0f));
      aSh[7] = (_Float16)(c2a * x * z);
    } else {                               // K slot 8 = feat 8
      aSh[0] = (_Float16)(c2c * (x * x - y * y));
    }

    // layer 1: (16x32) x (32x64) -> silu -> transpose to A-layout via LDS
#pragma unroll
    for (int t = 0; t < 4; ++t) {
      float bb = b1[t * 16 + c16];
      v8f acc;
#pragma unroll
      for (int r = 0; r < 8; ++r) acc[r] = bb;
      acc = wmma_f32_f16(aSh, load_a_f16(sW1t, 32, t * 16, 0, lane), acc);
#pragma unroll
      for (int r = 0; r < 8; ++r) {
        float xx = acc[r];
        xx = xx / (1.0f + expf(-xx));      // silu
        tp[(r + 8 * half) * 64 + t * 16 + c16] = (_Float16)xx;
      }
    }
    asm volatile("s_wait_dscnt 0" ::: "memory");
    v16h a20 = load_a_f16(tp, 64, 0, 0, lane);
    v16h a21 = load_a_f16(tp, 64, 0, 32, lane);

    // layer 2: (16x64) x (64x64) -> silu -> transpose again
#pragma unroll
    for (int t = 0; t < 4; ++t) {
      float bb = b2[t * 16 + c16];
      v8f acc;
#pragma unroll
      for (int r = 0; r < 8; ++r) acc[r] = bb;
      acc = wmma_f32_f16(a20, load_a_f16(sW2t, 64, t * 16, 0,  lane), acc);
      acc = wmma_f32_f16(a21, load_a_f16(sW2t, 64, t * 16, 32, lane), acc);
#pragma unroll
      for (int r = 0; r < 8; ++r) {
        float xx = acc[r];
        xx = xx / (1.0f + expf(-xx));
        tp[(r + 8 * half) * 64 + t * 16 + c16] = (_Float16)xx;
      }
    }
    asm volatile("s_wait_dscnt 0" ::: "memory");
    v16h a30 = load_a_f16(tp, 64, 0, 0, lane);
    v16h a31 = load_a_f16(tp, 64, 0, 32, lane);

    // layer 3: (16x64) x (64x16-padded), cols 0..7 = heads
    v8f acc3;
    {
      float bb = (c16 < 8) ? b3[c16] : 0.0f;
#pragma unroll
      for (int r = 0; r < 8; ++r) acc3[r] = bb;
    }
    acc3 = wmma_f32_f16(a30, load_a_f16(sW3t, 64, 0, 0,  lane), acc3);
    acc3 = wmma_f32_f16(a31, load_a_f16(sW3t, 64, 0, 32, lane), acc3);

    if (c16 < 8) {
      float* bp = bias + ((size_t)(b * H_ + c16) * S_ + i) * S_ + j0 + 8 * half;
#pragma unroll
      for (int r = 0; r < 8; ++r) bp[r] = acc3[r];   // 8 consecutive -> b128 x2
    }
  }
}

// ---- 4) flash attention: per-wave 16-query tile, Bc=32, no barriers --------
__global__ void attention_kernel(const _Float16* __restrict__ qf,
                                 const _Float16* __restrict__ kf,
                                 const _Float16* __restrict__ vT,
                                 const float* __restrict__ bias,
                                 _Float16* __restrict__ attn_out) {
  __shared__ _Float16 sP[4][16 * 32];     // per-wave P transpose bounce
  const int tid  = threadIdx.x;
  const int lane = tid & 31;
  const int w    = tid >> 5;
  const int half = lane >> 4;
  const int c16  = lane & 15;

  const int gid = blockIdx.x * 4 + w;     // 0..511 = (b,h,i-tile)
  const int it = gid & 31;
  const int h  = (gid >> 5) & 7;
  const int b  = gid >> 8;
  const int i0 = it * 16;

  const _Float16* qb  = qf + (size_t)(b * H_ + h) * S_ * DH_;
  const _Float16* kb  = kf + (size_t)(b * H_ + h) * S_ * DH_;
  const _Float16* vtb = vT + (size_t)(b * H_ + h) * DH_ * S_;   // [dh][s]
  const float*    bp  = bias + (size_t)(b * H_ + h) * S_ * S_;

  v16h aq0 = load_a_f16(qb, DH_, i0, 0, lane);
  v16h aq1 = load_a_f16(qb, DH_, i0, 32, lane);

  v8f o0, o1, o2, o3;
#pragma unroll
  for (int r = 0; r < 8; ++r) { o0[r] = 0.f; o1[r] = 0.f; o2[r] = 0.f; o3[r] = 0.f; }
  float m[8], l[8], alpha[8];
#pragma unroll
  for (int r = 0; r < 8; ++r) { m[r] = -1e30f; l[r] = 0.f; }

  _Float16* tp = &sP[w][0];

  for (int j0 = 0; j0 < S_; j0 += 32) {
    v8f s0, s1;
#pragma unroll
    for (int r = 0; r < 8; ++r) { s0[r] = 0.f; s1[r] = 0.f; }
    // B-layout of K^T == A-layout load of K (free transpose)
    s0 = wmma_f32_f16(aq0, load_a_f16(kb, DH_, j0,      0,  lane), s0);
    s0 = wmma_f32_f16(aq1, load_a_f16(kb, DH_, j0,      32, lane), s0);
    s1 = wmma_f32_f16(aq0, load_a_f16(kb, DH_, j0 + 16, 0,  lane), s1);
    s1 = wmma_f32_f16(aq1, load_a_f16(kb, DH_, j0 + 16, 32, lane), s1);
    if (j0 + 32 < S_)
      __builtin_prefetch(&kb[(size_t)(j0 + 32) * DH_], 0, 1);

#pragma unroll
    for (int r = 0; r < 8; ++r) {         // add SH bias (q pre-scaled by 1/8)
      int qi = i0 + r + 8 * half;
      s0[r] += bp[(size_t)qi * S_ + j0 + c16];
      s1[r] += bp[(size_t)qi * S_ + j0 + 16 + c16];
    }

#pragma unroll
    for (int r = 0; r < 8; ++r) {         // online softmax per query row
      float v = fmaxf(s0[r], s1[r]);
#pragma unroll
      for (int off = 8; off >= 1; off >>= 1) v = fmaxf(v, __shfl_xor(v, off));
      float mn = fmaxf(m[r], v);
      float a  = expf(m[r] - mn);
      float p0 = expf(s0[r] - mn);
      float p1 = expf(s1[r] - mn);
      float ps = p0 + p1;
#pragma unroll
      for (int off = 8; off >= 1; off >>= 1) ps += __shfl_xor(ps, off);
      l[r] = l[r] * a + ps;
      m[r] = mn;
      alpha[r] = a;
      tp[(r + 8 * half) * 32 + c16]      = (_Float16)p0;   // P -> A-layout via LDS
      tp[(r + 8 * half) * 32 + 16 + c16] = (_Float16)p1;
    }
#pragma unroll
    for (int r = 0; r < 8; ++r) {
      o0[r] *= alpha[r]; o1[r] *= alpha[r]; o2[r] *= alpha[r]; o3[r] *= alpha[r];
    }
    asm volatile("s_wait_dscnt 0" ::: "memory");
    v16h aP = load_a_f16(tp, 32, 0, 0, lane);
    // PV: B[k=key][n=dh]; vT is [dh][s] so B-tile == A-load of vT
    o0 = wmma_f32_f16(aP, load_a_f16(vtb, S_, 0,  j0, lane), o0);
    o1 = wmma_f32_f16(aP, load_a_f16(vtb, S_, 16, j0, lane), o1);
    o2 = wmma_f32_f16(aP, load_a_f16(vtb, S_, 32, j0, lane), o2);
    o3 = wmma_f32_f16(aP, load_a_f16(vtb, S_, 48, j0, lane), o3);
  }

#pragma unroll
  for (int r = 0; r < 8; ++r) {           // normalize; store f16 [B,S,D], d=h*64+dh
    int s = i0 + r + 8 * half;
    float inv = 1.0f / l[r];
    _Float16* op = attn_out + (size_t)(b * S_ + s) * D_ + h * DH_;
    op[ 0 + c16] = (_Float16)(o0[r] * inv);
    op[16 + c16] = (_Float16)(o1[r] * inv);
    op[32 + c16] = (_Float16)(o2[r] * inv);
    op[48 + c16] = (_Float16)(o3[r] * inv);
  }
}

// ---- 5) output projection: out = attn_out @ Wo + bo (transposed Wo) --------
__global__ void out_proj_kernel(const _Float16* __restrict__ attn,
                                const _Float16* __restrict__ WoT,
                                const float* __restrict__ bo,
                                float* __restrict__ out) {
  const int tid  = threadIdx.x;
  const int lane = tid & 31;
  const int w    = tid >> 5;
  const int half = lane >> 4;
  const int c16  = lane & 15;
  const int col0 = (blockIdx.x * 4 + w) * 16;
  const int row0 = blockIdx.y * 16;

  v8f acc;
  {
    float bb = bo[col0 + c16];
#pragma unroll
    for (int r = 0; r < 8; ++r) acc[r] = bb;
  }
#pragma unroll 4
  for (int k0 = 0; k0 < D_; k0 += 32) {
    acc = wmma_f32_f16(load_a_f16(attn, D_, row0, k0, lane),
                       load_a_f16(WoT, D_, col0, k0, lane), acc);
    if (k0 + 32 < D_)
      __builtin_prefetch(&WoT[(size_t)col0 * D_ + k0 + 32], 0, 1);
  }
#pragma unroll
  for (int r = 0; r < 8; ++r)
    out[(size_t)(row0 + r + 8 * half) * D_ + col0 + c16] = acc[r];
}

// ---- launch ----------------------------------------------------------------
extern "C" void kernel_launch(void* const* d_in, const int* in_sizes, int n_in,
                              void* d_out, int out_size, void* d_ws, size_t ws_size,
                              hipStream_t stream) {
  const float* q_data   = (const float*)d_in[0];
  const float* kv_feats = (const float*)d_in[1];
  const float* coords   = (const float*)d_in[2];
  const float* Wq = (const float*)d_in[3];
  const float* bq = (const float*)d_in[4];
  const float* Wk = (const float*)d_in[5];
  const float* bk = (const float*)d_in[6];
  const float* Wv = (const float*)d_in[7];
  const float* bv = (const float*)d_in[8];
  const float* Wo = (const float*)d_in[9];
  const float* bo = (const float*)d_in[10];
  const float* fw = (const float*)d_in[11];
  const float* offset = (const float*)d_in[12];
  const float* W1 = (const float*)d_in[13];
  const float* b1 = (const float*)d_in[14];
  const float* W2 = (const float*)d_in[15];
  const float* b2 = (const float*)d_in[16];
  const float* W3 = (const float*)d_in[17];
  const float* b3 = (const float*)d_in[18];

  char* ws = (char*)d_ws;
  _Float16* q16  = (_Float16*)ws; ws += (size_t)BS_ * D_ * sizeof(_Float16);
  _Float16* kv16 = (_Float16*)ws; ws += (size_t)BS_ * D_ * sizeof(_Float16);
  _Float16* WqT  = (_Float16*)ws; ws += (size_t)D_ * D_ * sizeof(_Float16);
  _Float16* WkT  = (_Float16*)ws; ws += (size_t)D_ * D_ * sizeof(_Float16);
  _Float16* WvT  = (_Float16*)ws; ws += (size_t)D_ * D_ * sizeof(_Float16);
  _Float16* WoT  = (_Float16*)ws; ws += (size_t)D_ * D_ * sizeof(_Float16);
  _Float16* qf   = (_Float16*)ws; ws += (size_t)BS_ * D_ * sizeof(_Float16);
  _Float16* kf   = (_Float16*)ws; ws += (size_t)BS_ * D_ * sizeof(_Float16);
  _Float16* vT   = (_Float16*)ws; ws += (size_t)BS_ * D_ * sizeof(_Float16);
  float*    bias = (float*)ws;    ws += (size_t)B_ * H_ * S_ * S_ * sizeof(float);
  _Float16* ao   = (_Float16*)ws; // attn output f16, BS_*D_

  const int WN = D_ * D_;   // 262144
  f32_to_f16_kernel<<<dim3(BS_ * D_ / 256), 256, 0, stream>>>(q_data, q16, BS_ * D_);
  transpose_conv_kernel<<<dim3(WN / 256), 256, 0, stream>>>(Wq, WqT);
  transpose_conv_kernel<<<dim3(WN / 256), 256, 0, stream>>>(Wk, WkT);
  transpose_conv_kernel<<<dim3(WN / 256), 256, 0, stream>>>(Wv, WvT);
  transpose_conv_kernel<<<dim3(WN / 256), 256, 0, stream>>>(Wo, WoT);

  kv_combine_kernel<<<dim3(BS_ * D_ / 256), 256, 0, stream>>>(kv_feats, fw, kv16);
  proj_qkv_kernel<<<dim3(8, 64, 3), dim3(128), 0, stream>>>(
      q16, kv16, WqT, bq, WkT, bk, WvT, bv, offset, qf, kf, vT);
  sh_bias_kernel<<<dim3(B_ * S_), dim3(128), 0, stream>>>(
      coords, W1, b1, W2, b2, W3, b3, bias);
  attention_kernel<<<dim3(128), dim3(128), 0, stream>>>(qf, kf, vT, bias, ao);
  out_proj_kernel<<<dim3(8, 64), dim3(128), 0, stream>>>(ao, WoT, bo, (float*)d_out);
}
